// SynthesisLayer_11888469476282
// MI455X (gfx1250) — matrix operand order repaired and checked by reading, compiled, not run on hardware
//
#include <hip/hip_runtime.h>

// ---------------------------------------------------------------------------
// StyleGAN3 SynthesisLayer for MI455X (gfx1250, wave32, WMMA + async-to-LDS)
//   B=16, CIN=COUT=256, H=W=64, K=3, up=down=2, taps=12
// ---------------------------------------------------------------------------

typedef __attribute__((ext_vector_type(4)))  unsigned int u32x4;
typedef __attribute__((ext_vector_type(16))) __bf16       v16bf;
typedef __attribute__((ext_vector_type(8)))  float        v8f;

union Frag { u32x4 u[2]; v16bf v; };

// async-to-LDS builtin operand types: v4i32 pointers in AS(1)/AS(3)
// (signature revealed by compiler diagnostic: first param is
//  '__attribute__((__vector_size__(4 * sizeof(int)))) int __device__ *')
typedef int v4i_vs __attribute__((vector_size(16)));
typedef __attribute__((address_space(1))) v4i_vs gv4i;
typedef __attribute__((address_space(3))) v4i_vs lv4i;

// s_wait_asynccnt: builtin if the toolchain has it, else raw asm
#if __has_builtin(__builtin_amdgcn_s_wait_asynccnt)
#define WAIT_ASYNC(n) __builtin_amdgcn_s_wait_asynccnt(n)
#else
#define WAIT_ASYNC(n) asm volatile("s_wait_asynccnt %0" ::"i"(n) : "memory")
#endif

// Per-lane async copy of 16 bytes global -> LDS (ASYNCcnt-tracked).
__device__ __forceinline__ void async_copy_b128(const void* gsrc, unsigned lds_byte_off) {
#if __has_builtin(__builtin_amdgcn_global_load_async_to_lds_b128)
    __builtin_amdgcn_global_load_async_to_lds_b128(
        (gv4i*)(size_t)gsrc,
        (lv4i*)(unsigned long)lds_byte_off,
        0, 0);
#else
    asm volatile("global_load_async_to_lds_b128 %0, %1, off"
                 :: "v"(lds_byte_off), "v"((unsigned long long)(size_t)gsrc)
                 : "memory");
#endif
}

// firwin(12, cutoff=16, width=32, fs=128) with Kaiser window (beta~4.2492),
// normalized to unit DC gain. Symmetric.
__device__ __constant__ float FILT[12] = {
    -0.0038537f, -0.0061462f,  0.0153733f,  0.0796452f,  0.1723545f, 0.2426267f,
     0.2426267f,  0.1723545f,  0.0796452f,  0.0153733f, -0.0061462f, -0.0038537f };

__device__ __forceinline__ unsigned short f2bf(float f) {
    unsigned int u = __float_as_uint(f);
    u += 0x7FFFu + ((u >> 16) & 1u);       // round-to-nearest-even
    return (unsigned short)(u >> 16);
}

// --- workspace layout (bytes) ----------------------------------------------
static constexpr size_t OFF_S    = 0;         // s_scaled [16*256] f32
static constexpr size_t OFF_WSC  = 16384;     // wt_scale [256] f32
static constexpr size_t OFF_G    = 17408;     // input_gain [1] f32
static constexpr size_t OFF_WMOD = 17920;     // wmod [16*256*2304] bf16, K=(kk)*256+cin
static constexpr size_t OFF_XT   = 18892288;  // xT [16*64*64*256] bf16 channel-last
static constexpr size_t OFF_Y0   = 52446720;  // y0 [16*256*66*66] f32

// ===========================================================================
// Kernel 1: affine styles + global style norm + per-cout weight norm + gain
// ===========================================================================
__global__ void __launch_bounds__(256)
prep_kernel(const float* __restrict__ w, const float* __restrict__ aw,
            const float* __restrict__ ab, const float* __restrict__ cw,
            const float* __restrict__ ema,
            float* __restrict__ s_scaled, float* __restrict__ wt_scale,
            float* __restrict__ gainp)
{
    const int tid = threadIdx.x;
    const float scale = 0.04419417382415922f;    // 1/sqrt(512)
    float st[16];
    float sumsq = 0.f;
    const float* ar = aw + (size_t)tid * 512;
    for (int b = 0; b < 16; ++b) {
        const float* wr = w + b * 512;
        float acc = 0.f;
        for (int k = 0; k < 512; ++k) acc += wr[k] * ar[k];
        acc = acc * scale + ab[tid];
        st[b] = acc;
        sumsq += acc * acc;
    }
    __shared__ float red[256];
    red[tid] = sumsq; __syncthreads();
    for (int s = 128; s > 0; s >>= 1) {
        if (tid < s) red[tid] += red[tid + s];
        __syncthreads();
    }
    const float rs = rsqrtf(red[0] * (1.0f / 4096.0f));
    for (int b = 0; b < 16; ++b) s_scaled[b * 256 + tid] = st[b] * rs;

    const float* cr = cw + (size_t)tid * 2304;
    float a2 = 0.f;
    for (int k = 0; k < 2304; ++k) a2 += cr[k] * cr[k];
    wt_scale[tid] = rsqrtf(a2 * (1.0f / 2304.0f));

    if (tid == 0) gainp[0] = rsqrtf(ema[0]);
}

// ===========================================================================
// Kernel 2: modulate + demodulate -> bf16 weights, K reordered (kk,cin)
// ===========================================================================
__global__ void __launch_bounds__(256)
modulate_kernel(const float* __restrict__ cw, const float* __restrict__ s_scaled,
                const float* __restrict__ wt_scale, const float* __restrict__ gainp,
                unsigned short* __restrict__ wmod)
{
    const int b = blockIdx.x >> 8;
    const int o = blockIdx.x & 255;
    const int tid = threadIdx.x;
    const float wsc = wt_scale[o];

    float vals[9];
    float sum = 0.f;
    #pragma unroll
    for (int k = 0; k < 9; ++k) {
        const int j = k * 256 + tid;
        const int i = j / 9;
        const float v = cw[(size_t)o * 2304 + j] * wsc * s_scaled[b * 256 + i];
        vals[k] = v;
        sum += v * v;
    }
    __shared__ float red[256];
    red[tid] = sum; __syncthreads();
    for (int s = 128; s > 0; s >>= 1) {
        if (tid < s) red[tid] += red[tid + s];
        __syncthreads();
    }
    const float d = rsqrtf(red[0] + 1e-8f) * gainp[0];
    unsigned short* dst = wmod + ((size_t)(b * 256 + o)) * 2304;
    #pragma unroll
    for (int k = 0; k < 9; ++k) {
        const int j = k * 256 + tid;
        const int i = j / 9;
        const int kk = j - i * 9;
        dst[kk * 256 + i] = f2bf(vals[k] * d);
    }
}

// ===========================================================================
// Kernel 3: x (NCHW f32) -> channel-last bf16 xT[b][y][x][c]
// ===========================================================================
__global__ void __launch_bounds__(256)
xpose_kernel(const float* __restrict__ x, unsigned short* __restrict__ xT)
{
    const int bid = blockIdx.x;
    const int b = bid >> 12, y = (bid >> 6) & 63, xc = bid & 63;
    const int c = threadIdx.x;
    const float v = x[(((size_t)(b * 256 + c)) * 64 + y) * 64 + xc];
    xT[(((size_t)(b * 64 + y)) * 64 + xc) * 256 + c] = f2bf(v);
}

// ===========================================================================
// Kernel 4: implicit-GEMM 3x3 conv via v_wmma_f32_16x16x32_bf16
//   One workgroup (8 waves) per (b, 64-pixel block). The im2col B-tile
//   (64 pix x 32 K bf16, rows padded to 80B) is staged in double-buffered LDS
//   via GLOBAL_LOAD_ASYNC_TO_LDS_B128 (1 issue per wave per K-chunk), waited
//   with s_wait_asynccnt, and shared by all 8 waves. Each wave owns
//   2 cout-tiles x 4 pixel-tiles = 8 WMMA accumulators.
//   72 K-chunks: chunk c -> tap kkpos=c>>3, cin block i0=(c&7)*32.
// grid = 16 b * 69 pt = 1104 blocks, 256 threads
// ===========================================================================
#define BT_ROW   80        // padded LDS row stride (64B data + 16B pad)
#define BT_SIZE  (64 * BT_ROW)

__global__ void __launch_bounds__(256)
conv_wmma_kernel(const unsigned short* __restrict__ wmod,
                 const unsigned short* __restrict__ xT,
                 const float* __restrict__ bias,
                 float* __restrict__ y0)
{
    __shared__ unsigned char btile[2][BT_SIZE];

    const int tid  = threadIdx.x;
    const int lane = tid & 31;
    const int wave = tid >> 5;
    const int b  = blockIdx.x / 69;
    const int pt = blockIdx.x - b * 69;
    const int p0 = pt * 64;

    // ---- copy-role decomposition: thread t moves 16B for pixel t>>2, quarter t&3
    const int pixT = tid >> 2;
    const int qT   = tid & 3;
    const int pT   = p0 + pixT;
    const bool pvT = (pT < 4356);
    const int ppT  = pvT ? pT : 0;
    const int oyT  = ppT / 66;
    const int oxT  = ppT - oyT * 66;
    const unsigned lds_base = (unsigned)(size_t)&btile[0][0];

    auto issue_copy = [&](int c, int nb) {
        const int kkpos = c >> 3;
        const int i0 = (c & 7) * 32;
        const int dy = kkpos / 3;
        const int dx = kkpos - dy * 3;
        const int iy = oyT + dy - 2;
        const int ix = oxT + dx - 2;
        const bool ok = pvT & ((unsigned)iy < 64u) & ((unsigned)ix < 64u);
        const unsigned short* g = ok
            ? xT + (((size_t)((b * 64 + iy) * 64 + ix)) * 256 + i0 + qT * 8)
            : xT;                                  // safe dummy; masked at use
        async_copy_b128(g, lds_base + (unsigned)(nb * BT_SIZE + pixT * BT_ROW + qT * 16));
    };

    // ---- compute-role decomposition
    const int n   = lane & 15;            // A row (M) / B col (N = pixel)
    const int kbA = (lane >> 4) * 8;      // A: K in {kb..kb+7, kb+16..kb+23}
    const int kbB = (lane >> 4) * 8;      // B: K in {kb..kb+7, kb+16..kb+23}
    const int cout0 = wave * 32;

    const unsigned short* Arow0 = wmod + ((size_t)(b * 256 + cout0 + n)) * 2304;
    const unsigned short* Arow1 = Arow0 + (size_t)16 * 2304;

    int oy[4], ox[4]; bool pv[4];
    #pragma unroll
    for (int s = 0; s < 4; ++s) {
        const int p = p0 + s * 16 + n;
        pv[s] = (p < 4356);
        const int pp = pv[s] ? p : 0;
        oy[s] = pp / 66;
        ox[s] = pp - oy[s] * 66;
    }

    v8f acc0[4], acc1[4];
    #pragma unroll
    for (int s = 0; s < 4; ++s) { v8f z = {}; acc0[s] = z; acc1[s] = z; }

    issue_copy(0, 0);                     // prologue: fill buffer 0

    for (int c = 0; c < 72; ++c) {
        const int cb = c & 1;
        if (c + 1 < 72) {
            issue_copy(c + 1, cb ^ 1);    // prefetch next chunk
            WAIT_ASYNC(1);                // oldest (current buffer) complete
        } else {
            WAIT_ASYNC(0);
        }
        __syncthreads();                  // all waves' copies visible

        const int kkpos = c >> 3;
        const int dy = kkpos / 3;
        const int dx = kkpos - dy * 3;

        Frag a0, a1;
        a0.u[0] = *(const u32x4*)(Arow0 + c * 32 + kbA);
        a0.u[1] = *(const u32x4*)(Arow0 + c * 32 + kbA + 16);
        a1.u[0] = *(const u32x4*)(Arow1 + c * 32 + kbA);
        a1.u[1] = *(const u32x4*)(Arow1 + c * 32 + kbA + 16);
        if (c + 1 < 72) __builtin_prefetch(Arow0 + (c + 1) * 32, 0, 1);

        Frag bf[4];
        const unsigned char* bufp = btile[cb];
        #pragma unroll
        for (int s = 0; s < 4; ++s) {
            const u32x4* sp = (const u32x4*)(bufp + (s * 16 + n) * BT_ROW + kbB * 2);
            bf[s].u[0] = sp[0];
            bf[s].u[1] = sp[2];
            const int iy = oy[s] + dy - 2;
            const int ix = ox[s] + dx - 2;
            if (!(pv[s] & ((unsigned)iy < 64u) & ((unsigned)ix < 64u))) {
                bf[s].u[0] = (u32x4){0u, 0u, 0u, 0u};
                bf[s].u[1] = (u32x4){0u, 0u, 0u, 0u};
            }
        }
        __syncthreads();                  // LDS reads done before next overwrite

        #pragma unroll
        for (int s = 0; s < 4; ++s) {
            acc0[s] = __builtin_amdgcn_wmma_f32_16x16x32_bf16(
                false, a0.v, false, bf[s].v, (short)0, acc0[s], false, false);
            acc1[s] = __builtin_amdgcn_wmma_f32_16x16x32_bf16(
                false, a1.v, false, bf[s].v, (short)0, acc1[s], false, false);
        }
    }

    // C/D layout: VGPR r, lanes 0-15 -> M=r, lanes 16-31 -> M=r+8, N=lane&15
    const int mofs = (lane >> 4) * 8;
    #pragma unroll
    for (int s = 0; s < 4; ++s) {
        const int p = p0 + s * 16 + n;
        if (p < 4356) {
            #pragma unroll
            for (int r = 0; r < 8; ++r) {
                const int o0 = cout0 + mofs + r;
                y0[((size_t)(b * 256 + o0)) * 4356 + p] = acc0[s][r] + bias[o0];
                const int o1 = o0 + 16;
                y0[((size_t)(b * 256 + o1)) * 4356 + p] = acc1[s][r] + bias[o1];
            }
        }
    }
}

// ===========================================================================
// Kernel 5: fused filtered_lrelu, one (b,cout) channel per block in big LDS.
//   66x66 -> (v-up 2x FIR) 138x66 -> (h-up) 138x138 -> lrelu*sqrt2, clamp
//   -> (v-down stride2) 64x138 -> (h-down stride2) 64x64
// dynamic LDS = (4356 + 9108 + 19044) * 4 = 130032 bytes (<< 320KB/WGP)
// ===========================================================================
__global__ void __launch_bounds__(256)
flrelu_kernel(const float* __restrict__ y0, float* __restrict__ out)
{
    extern __shared__ float smem[];
    float* ly0 = smem;                 // 66*66  = 4356
    float* up1 = smem + 4356;          // 138*66 = 9108  (reused as 64*138 dv)
    float* up2 = smem + 4356 + 9108;   // 138*138= 19044

    const int tid = threadIdx.x;
    const int bo  = blockIdx.x;
    const float* src = y0 + (size_t)bo * 4356;

    for (int i = tid; i < 4356; i += 256) ly0[i] = src[i];
    __syncthreads();

    for (int idx = tid; idx < 138 * 66; idx += 256) {
        const int r = idx / 66, col = idx - (idx / 66) * 66;
        float acc = 0.f;
        #pragma unroll
        for (int t = 0; t < 12; ++t) {
            const int d = r + t - 9;
            if (((d & 1) == 0) && d >= 0 && (d >> 1) < 66)
                acc += FILT[t] * ly0[(d >> 1) * 66 + col];
        }
        up1[idx] = acc * 2.0f;
    }
    __syncthreads();

    for (int idx = tid; idx < 138 * 138; idx += 256) {
        const int r = idx / 138, cc = idx - (idx / 138) * 138;
        float acc = 0.f;
        #pragma unroll
        for (int t = 0; t < 12; ++t) {
            const int d = cc + t - 9;
            if (((d & 1) == 0) && d >= 0 && (d >> 1) < 66)
                acc += FILT[t] * up1[r * 66 + (d >> 1)];
        }
        float v = acc * 2.0f;
        v = (v >= 0.f ? v : 0.2f * v) * 1.4142135623730951f;
        v = fminf(fmaxf(v, -256.f), 256.f);
        up2[idx] = v;
    }
    __syncthreads();

    for (int idx = tid; idx < 64 * 138; idx += 256) {
        const int R = idx / 138, c = idx - (idx / 138) * 138;
        float acc = 0.f;
        #pragma unroll
        for (int t = 0; t < 12; ++t)
            acc += FILT[t] * up2[(2 * R + t) * 138 + c];
        up1[idx] = acc;
    }
    __syncthreads();

    float* dst = out + (size_t)bo * 4096;
    for (int idx = tid; idx < 64 * 64; idx += 256) {
        const int R = idx >> 6, C = idx & 63;
        float acc = 0.f;
        #pragma unroll
        for (int t = 0; t < 12; ++t)
            acc += FILT[t] * up1[R * 138 + 2 * C + t];
        dst[idx] = acc;
    }
}

// ===========================================================================
extern "C" void kernel_launch(void* const* d_in, const int* in_sizes, int n_in,
                              void* d_out, int out_size, void* d_ws, size_t ws_size,
                              hipStream_t stream)
{
    const float* x   = (const float*)d_in[0];
    const float* w   = (const float*)d_in[1];
    const float* aw  = (const float*)d_in[2];
    const float* ab  = (const float*)d_in[3];
    const float* cw  = (const float*)d_in[4];
    const float* cb  = (const float*)d_in[5];
    const float* ema = (const float*)d_in[6];
    float* out = (float*)d_out;

    char* ws = (char*)d_ws;
    float*          s_scaled = (float*)(ws + OFF_S);
    float*          wt_scale = (float*)(ws + OFF_WSC);
    float*          gainp    = (float*)(ws + OFF_G);
    unsigned short* wmod     = (unsigned short*)(ws + OFF_WMOD);
    unsigned short* xT       = (unsigned short*)(ws + OFF_XT);
    float*          y0       = (float*)(ws + OFF_Y0);

    prep_kernel<<<1, 256, 0, stream>>>(w, aw, ab, cw, ema, s_scaled, wt_scale, gainp);
    modulate_kernel<<<4096, 256, 0, stream>>>(cw, s_scaled, wt_scale, gainp, wmod);
    xpose_kernel<<<65536, 256, 0, stream>>>(x, xT);
    conv_wmma_kernel<<<1104, 256, 0, stream>>>(wmod, xT, cb, y0);
    flrelu_kernel<<<4096, 256, 130032, stream>>>(y0, out);
}
